// CausalSelfAttention_19198503813853
// MI455X (gfx1250) — matrix operand order repaired and checked by reading, not compiled, not run on hardware
//
#include <hip/hip_runtime.h>

#define T_SEQ  2048
#define D_EMB  4096
#define QKV_O  6144
#define NHEAD  32
#define NGRP   8
#define HEADD  128
#define WIN    1024

typedef __attribute__((ext_vector_type(16))) __bf16 v16bf;
typedef __attribute__((ext_vector_type(8)))  float  v8f;
typedef unsigned int u32x4 __attribute__((ext_vector_type(4)));
typedef int          i32x8 __attribute__((ext_vector_type(8)));
typedef int          i32x4 __attribute__((ext_vector_type(4)));

#if __has_builtin(__builtin_amdgcn_tensor_load_to_lds)
#define HAVE_TDM 1
#else
#define HAVE_TDM 0
#endif

__device__ __forceinline__ unsigned short f2bf(float f) {
  unsigned int u = __float_as_uint(f);
  u += 0x7fffu + ((u >> 16) & 1u);          // round-to-nearest-even
  return (unsigned short)(u >> 16);
}
__device__ __forceinline__ unsigned int pk2(float a, float b) {
  return (unsigned int)f2bf(a) | ((unsigned int)f2bf(b) << 16);
}
__device__ __forceinline__ uint4 pack8(const float4& a, const float4& b) {
  uint4 r; r.x = pk2(a.x, a.y); r.y = pk2(a.z, a.w);
  r.z = pk2(b.x, b.y); r.w = pk2(b.z, b.w); return r;
}
__device__ __forceinline__ v16bf mkfrag(uint4 lo, uint4 hi) {
  union { uint4 u[2]; v16bf v; } x; x.u[0] = lo; x.u[1] = hi; return x.v;
}
__device__ __forceinline__ v8f wmma_bf16(v16bf a, v16bf b, v8f c) {
  return __builtin_amdgcn_wmma_f32_16x16x32_bf16(false, a, false, b,
                                                 (short)0, c, false, false);
}

#if HAVE_TDM
// 2-D TDM tile load: tile_rows x tile_k bf16 elements from a [nrows, stride]
// bf16 tensor into LDS at lds_off, packed [tile_rows][tile_k] row-major.
// D# layout per CDNA5 ISA sec 8.3-8.5. 6-arg builtin form (clang-23).
__device__ __forceinline__ void tdm_load_2d(unsigned int lds_off,
                                            const void* gaddr, int rowlen,
                                            int nrows, int stride,
                                            int tile_k, int tile_rows) {
  unsigned long long ga = (unsigned long long)(uintptr_t)gaddr;
  u32x4 g0;
  g0[0] = 1u;                                  // count=1, user descriptor
  g0[1] = lds_off;                             // lds_addr (bytes)
  g0[2] = (unsigned int)ga;                    // global_addr[31:0]
  g0[3] = (unsigned int)(ga >> 32) | (2u << 30); // global_addr[56:32] | type=2
  i32x8 g1;
  g1[0] = (int)(1u << 16);                     // wg_mask=0, data_size=1 (2B)
  g1[1] = (int)(((unsigned)rowlen & 0xffffu) << 16);            // tensor_dim0 lo
  g1[2] = (int)((((unsigned)rowlen >> 16) & 0xffffu) |
                (((unsigned)nrows & 0xffffu) << 16));           // dim0 hi | dim1 lo
  g1[3] = (int)((((unsigned)nrows >> 16) & 0xffffu) |
                ((unsigned)tile_k << 16));                      // dim1 hi | tile_dim0
  g1[4] = (int)((unsigned)tile_rows);          // tile_dim1, tile_dim2=0
  g1[5] = (int)((unsigned)stride);             // tensor_dim0_stride[31:0]
  g1[6] = 0;                                   // stride hi, dim1_stride lo
  g1[7] = 0;
  i32x4 z4; z4[0] = 0; z4[1] = 0; z4[2] = 0; z4[3] = 0;
  i32x8 z8;
#pragma unroll
  for (int i = 0; i < 8; i++) z8[i] = 0;
  __builtin_amdgcn_tensor_load_to_lds(g0, g1, z4, z4, z8, 0);
}
#endif

// elementwise f32 -> bf16 (n multiple of 8)
__global__ __launch_bounds__(256) void cvt_bf16(const float* __restrict__ in,
                                                unsigned short* __restrict__ out,
                                                int n) {
  int i = (blockIdx.x * 256 + threadIdx.x) * 8;
  if (i < n) {
    const float4* p = (const float4*)(in + i);
    *(uint4*)(out + i) = pack8(p[0], p[1]);
  }
}

// C[M,N] = A[M,K] * Bw[N,K]^T, A/Bw bf16, C fp32. BK=64, double-buffered.
__global__ __launch_bounds__(256) void gemm_bf16(
    const unsigned short* __restrict__ A, const unsigned short* __restrict__ Bw,
    float* __restrict__ C, int M, int N, int K) {
  __shared__ __align__(16) unsigned short As[2][128 * 64];
  __shared__ __align__(16) unsigned short Bs[2][128 * 64];
  const int tid = threadIdx.x;
  const int lane = tid & 31, w = tid >> 5;
  const int wm = w & 1, wn = w >> 1;           // 2 x 4 wave grid
  const int half = lane >> 4, l16 = lane & 15;
  const int m0 = blockIdx.y * 128, n0 = blockIdx.x * 128;

  v8f acc[4][2];
#pragma unroll
  for (int i = 0; i < 4; i++)
#pragma unroll
    for (int j = 0; j < 2; j++)
#pragma unroll
      for (int e = 0; e < 8; e++) acc[i][j][e] = 0.0f;

  const int nstages = K >> 6;

#if HAVE_TDM
  if (w == 0) {
    tdm_load_2d((unsigned int)(uintptr_t)&As[0][0], A + (size_t)m0 * K,
                K, M, K, 64, 128);
    tdm_load_2d((unsigned int)(uintptr_t)&Bs[0][0], Bw + (size_t)n0 * K,
                K, N, K, 64, 128);
  }
#endif

  for (int s = 0; s < nstages; s++) {
    const int p = s & 1;
#if HAVE_TDM
    if (w == 0) {
      if (s + 1 < nstages) {
        const int k1 = (s + 1) << 6;
        tdm_load_2d((unsigned int)(uintptr_t)&As[1 - p][0],
                    A + (size_t)m0 * K + k1, K, M, K, 64, 128);
        tdm_load_2d((unsigned int)(uintptr_t)&Bs[1 - p][0],
                    Bw + (size_t)n0 * K + k1, K, N, K, 64, 128);
        __builtin_amdgcn_s_wait_tensorcnt((short)2);
      } else {
        __builtin_amdgcn_s_wait_tensorcnt((short)0);
      }
    }
#else
    {  // manual staging: 256 threads copy 128x64 bf16 per matrix
      const int k0 = s << 6;
      const int row = tid >> 1, seg = tid & 1;
      const uint4* ga = (const uint4*)(A + (size_t)(m0 + row) * K + k0 + seg * 32);
      const uint4* gb = (const uint4*)(Bw + (size_t)(n0 + row) * K + k0 + seg * 32);
      uint4* da = (uint4*)&As[p][row * 64 + seg * 32];
      uint4* db = (uint4*)&Bs[p][row * 64 + seg * 32];
#pragma unroll
      for (int c = 0; c < 4; c++) { da[c] = ga[c]; db[c] = gb[c]; }
    }
#endif
    __syncthreads();

    const unsigned short* Asp = &As[p][0];
    const unsigned short* Bsp = &Bs[p][0];
#pragma unroll
    for (int kk = 0; kk < 64; kk += 32) {
      v16bf af[4], bfr[2];
#pragma unroll
      for (int mi = 0; mi < 4; mi++) {
        const uint4* pa = (const uint4*)&Asp[(wm * 64 + mi * 16 + l16) * 64 + kk + half * 16];
        af[mi] = mkfrag(pa[0], pa[1]);
      }
#pragma unroll
      for (int ni = 0; ni < 2; ni++) {
        const uint4* pb = (const uint4*)&Bsp[(wn * 32 + ni * 16 + l16) * 64 + kk + half * 16];
        bfr[ni] = mkfrag(pb[0], pb[1]);
      }
#pragma unroll
      for (int mi = 0; mi < 4; mi++)
#pragma unroll
        for (int ni = 0; ni < 2; ni++)
          acc[mi][ni] = wmma_bf16(af[mi], bfr[ni], acc[mi][ni]);
    }
    __syncthreads();
  }

#pragma unroll
  for (int mi = 0; mi < 4; mi++)
#pragma unroll
    for (int ni = 0; ni < 2; ni++) {
      int gn = n0 + wn * 32 + ni * 16 + l16;
#pragma unroll
      for (int r = 0; r < 8; r++) {
        int gm = m0 + wm * 64 + mi * 16 + half * 8 + r;
        C[(size_t)gm * N + gn] = acc[mi][ni][r];
      }
    }
}

// in-place rmsnorm(4096) + per-head RoPE on q slice of qkv
__global__ __launch_bounds__(256) void rms_rope_q(
    float* __restrict__ qkv, const float* __restrict__ cs,
    const float* __restrict__ sn, const float* __restrict__ wq) {
  __shared__ float red[256];
  const int t = blockIdx.x, tid = threadIdx.x;
  float* q = qkv + (size_t)t * QKV_O;
  const int hh = tid >> 3, s8 = tid & 7;
  float a[8], b[8];
  float ss = 0.f;
#pragma unroll
  for (int i = 0; i < 8; i++) {
    int d = s8 * 8 + i;
    a[i] = q[hh * 128 + d];
    b[i] = q[hh * 128 + d + 64];
    ss += a[i] * a[i] + b[i] * b[i];
  }
  red[tid] = ss; __syncthreads();
#pragma unroll
  for (int off = 128; off >= 1; off >>= 1) {
    if (tid < off) red[tid] += red[tid + off];
    __syncthreads();
  }
  float inv = rsqrtf(red[0] * (1.0f / 4096.0f) + 1e-5f);
#pragma unroll
  for (int i = 0; i < 8; i++) {
    int d = s8 * 8 + i;
    float an = a[i] * inv * wq[hh * 128 + d];
    float bn = b[i] * inv * wq[hh * 128 + d + 64];
    float c1 = cs[t * 128 + d],      s1 = sn[t * 128 + d];
    float c2 = cs[t * 128 + d + 64], s2 = sn[t * 128 + d + 64];
    q[hh * 128 + d]      = an * c1 - bn * s1;
    q[hh * 128 + d + 64] = bn * c2 + an * s2;
  }
}

// in-place rmsnorm(1024) + per-group RoPE on k slice of qkv
__global__ __launch_bounds__(256) void rms_rope_k(
    float* __restrict__ qkv, const float* __restrict__ cs,
    const float* __restrict__ sn, const float* __restrict__ wk) {
  __shared__ float red[256];
  const int t = blockIdx.x, tid = threadIdx.x;
  float* k = qkv + (size_t)t * QKV_O + D_EMB;
  const int g = tid >> 5, i32 = tid & 31;
  float a[2], b[2];
  float ss = 0.f;
#pragma unroll
  for (int i = 0; i < 2; i++) {
    int d = i32 * 2 + i;
    a[i] = k[g * 128 + d];
    b[i] = k[g * 128 + d + 64];
    ss += a[i] * a[i] + b[i] * b[i];
  }
  red[tid] = ss; __syncthreads();
#pragma unroll
  for (int off = 128; off >= 1; off >>= 1) {
    if (tid < off) red[tid] += red[tid + off];
    __syncthreads();
  }
  float inv = rsqrtf(red[0] * (1.0f / 1024.0f) + 1e-5f);
#pragma unroll
  for (int i = 0; i < 2; i++) {
    int d = i32 * 2 + i;
    float an = a[i] * inv * wk[g * 128 + d];
    float bn = b[i] * inv * wk[g * 128 + d + 64];
    float c1 = cs[t * 128 + d],      s1 = sn[t * 128 + d];
    float c2 = cs[t * 128 + d + 64], s2 = sn[t * 128 + d + 64];
    k[g * 128 + d]      = an * c1 - bn * s1;
    k[g * 128 + d + 64] = bn * c2 + an * s2;
  }
}

// flash attention, sliding window 1024, GQA 4:1. One block = (head, 128-q tile).
// Writes y in bf16 for the projection GEMM.
__global__ __launch_bounds__(256) void attn_swin(
    const float* __restrict__ qkv, unsigned short* __restrict__ y) {
  __shared__ __align__(16) unsigned short Qs[128 * 128];   // [q][d]
  __shared__ __align__(16) unsigned short Ks[32 * 128];    // [key][d]
  __shared__ __align__(16) unsigned short Vt[128 * 32];    // [d][key]
  __shared__ __align__(16) unsigned short Ps[8 * 16 * 32]; // per-wave P tiles

  const int h = blockIdx.x, qt = blockIdx.y, g = h >> 2;
  const int q0 = qt * 128;
  const int tid = threadIdx.x, lane = tid & 31, w = tid >> 5;
  const int half = lane >> 4, l16 = lane & 15;

  {  // stage Q tile (128 x 128) f32 -> bf16
    int row = tid >> 1, seg = tid & 1;
    const float4* gq = (const float4*)(qkv + (size_t)(q0 + row) * QKV_O + h * HEADD + seg * 64);
    uint4* dst = (uint4*)&Qs[row * 128 + seg * 64];
#pragma unroll
    for (int c = 0; c < 8; c++) dst[c] = pack8(gq[2 * c], gq[2 * c + 1]);
  }

  v8f o[8];
#pragma unroll
  for (int f = 0; f < 8; f++)
#pragma unroll
    for (int e = 0; e < 8; e++) o[f][e] = 0.0f;
  float mrow[8], lrow[8];
#pragma unroll
  for (int r = 0; r < 8; r++) { mrow[r] = -3.0e38f; lrow[r] = 0.0f; }

  int jlo = q0 - (WIN - 1); if (jlo < 0) jlo = 0; jlo &= ~31;
  const float scale = 0.088388347648318447f;  // 1/sqrt(128)
  unsigned short* pw = &Ps[w * 512];

  for (int j0 = jlo; j0 < q0 + 128; j0 += 32) {
    {  // stage K (32x128) and V transposed (128x32)
      int row = tid >> 3, seg = tid & 7;
      const float4* gk = (const float4*)(qkv + (size_t)(j0 + row) * QKV_O + D_EMB + g * HEADD + seg * 16);
      float4 k0f = gk[0], k1f = gk[1], k2f = gk[2], k3f = gk[3];
      uint4* dk = (uint4*)&Ks[row * 128 + seg * 16];
      dk[0] = pack8(k0f, k1f); dk[1] = pack8(k2f, k3f);
      const float* gv = qkv + (size_t)(j0 + row) * QKV_O + D_EMB + NGRP * HEADD + g * HEADD + seg * 16;
#pragma unroll
      for (int e = 0; e < 16; e++) Vt[(seg * 16 + e) * 32 + row] = f2bf(gv[e]);
    }
    __syncthreads();

    // S = Q * K^T  (16 q-rows per wave x 32 keys)
    v8f s0, s1;
#pragma unroll
    for (int e = 0; e < 8; e++) { s0[e] = 0.0f; s1[e] = 0.0f; }
#pragma unroll
    for (int kk = 0; kk < 128; kk += 32) {
      const uint4* pa = (const uint4*)&Qs[(w * 16 + l16) * 128 + kk + half * 16];
      v16bf aq = mkfrag(pa[0], pa[1]);
      const uint4* pb0 = (const uint4*)&Ks[l16 * 128 + kk + half * 16];
      const uint4* pb1 = (const uint4*)&Ks[(16 + l16) * 128 + kk + half * 16];
      s0 = wmma_bf16(aq, mkfrag(pb0[0], pb0[1]), s0);
      s1 = wmma_bf16(aq, mkfrag(pb1[0], pb1[1]), s1);
    }

    // mask + online softmax (row = half*8+r within wave's 16-row strip)
    float sc[8];
#pragma unroll
    for (int r = 0; r < 8; r++) {
      const int i = q0 + w * 16 + half * 8 + r;
      const int ja = j0 + l16, jb = ja + 16;
      float sa = ((ja <= i) && (i - ja < WIN)) ? s0[r] * scale : -3.0e38f;
      float sb = ((jb <= i) && (i - jb < WIN)) ? s1[r] * scale : -3.0e38f;
      float tmax = fmaxf(sa, sb);
#pragma unroll
      for (int m = 8; m >= 1; m >>= 1) tmax = fmaxf(tmax, __shfl_xor(tmax, m, 32));
      float mn = fmaxf(mrow[r], tmax);
      float mref = (mn > -1.0e37f) ? mn : 0.0f;
      float scv = __expf(mrow[r] - mref);
      float pa0 = __expf(sa - mref);
      float pb0v = __expf(sb - mref);
      float ps = pa0 + pb0v;
#pragma unroll
      for (int m = 8; m >= 1; m >>= 1) ps += __shfl_xor(ps, m, 32);
      lrow[r] = lrow[r] * scv + ps;
      mrow[r] = mn;
      sc[r] = scv;
      pw[(half * 8 + r) * 32 + l16]      = f2bf(pa0);
      pw[(half * 8 + r) * 32 + 16 + l16] = f2bf(pb0v);
    }
#pragma unroll
    for (int f = 0; f < 8; f++)
#pragma unroll
      for (int r = 0; r < 8; r++) o[f][r] *= sc[r];

    asm volatile("s_wait_dscnt 0" ::: "memory");  // P LDS round-trip fence

    {  // O += P * V
      const uint4* pa = (const uint4*)&pw[l16 * 32 + half * 16];
      v16bf ap = mkfrag(pa[0], pa[1]);
#pragma unroll
      for (int f = 0; f < 8; f++) {
        const uint4* pb = (const uint4*)&Vt[(f * 16 + l16) * 32 + half * 16];
        o[f] = wmma_bf16(ap, mkfrag(pb[0], pb[1]), o[f]);
      }
    }
    __syncthreads();
  }

  float invl[8];
#pragma unroll
  for (int r = 0; r < 8; r++) invl[r] = (lrow[r] > 0.f) ? 1.0f / lrow[r] : 0.0f;
#pragma unroll
  for (int f = 0; f < 8; f++)
#pragma unroll
    for (int r = 0; r < 8; r++) {
      int row = q0 + w * 16 + half * 8 + r;
      int col = h * HEADD + f * 16 + l16;
      y[(size_t)row * D_EMB + col] = f2bf(o[f][r] * invl[r]);
    }
}

extern "C" void kernel_launch(void* const* d_in, const int* in_sizes, int n_in,
                              void* d_out, int out_size, void* d_ws, size_t ws_size,
                              hipStream_t stream) {
  (void)in_sizes; (void)n_in; (void)out_size; (void)ws_size;
  const float* x        = (const float*)d_in[0];  // [1,2048,4096]
  const float* cosT     = (const float*)d_in[1];  // [2048,128]
  const float* sinT     = (const float*)d_in[2];  // [2048,128]
  const float* W_qkv    = (const float*)d_in[3];  // [6144,4096]
  const float* norm_q_w = (const float*)d_in[4];  // [4096]
  const float* norm_k_w = (const float*)d_in[5];  // [1024]
  const float* W_proj   = (const float*)d_in[6];  // [4096,4096]
  float* out = (float*)d_out;                     // [2048,4096]

  // workspace layout
  float* qkv = (float*)d_ws;                                    // 2048*6144 f32
  unsigned short* xy_bf = (unsigned short*)(qkv + (size_t)T_SEQ * QKV_O);  // x_bf then y_bf
  unsigned short* w_bf  = xy_bf + (size_t)T_SEQ * D_EMB;        // W_qkv_bf then W_proj_bf

  const int n_x  = T_SEQ * D_EMB;
  const int n_wq = QKV_O * D_EMB;
  const int n_wp = D_EMB * D_EMB;

  dim3 blk(256);
  cvt_bf16<<<n_x  / 2048, blk, 0, stream>>>(x, xy_bf, n_x);
  cvt_bf16<<<n_wq / 2048, blk, 0, stream>>>(W_qkv, w_bf, n_wq);
  gemm_bf16<<<dim3(QKV_O / 128, T_SEQ / 128), blk, 0, stream>>>(
      xy_bf, w_bf, qkv, T_SEQ, QKV_O, D_EMB);
  rms_rope_q<<<T_SEQ, 256, 0, stream>>>(qkv, cosT, sinT, norm_q_w);
  rms_rope_k<<<T_SEQ, 256, 0, stream>>>(qkv, cosT, sinT, norm_k_w);
  cvt_bf16<<<n_wp / 2048, blk, 0, stream>>>(W_proj, w_bf, n_wp);  // reuse buffer
  attn_swin<<<dim3(NHEAD, T_SEQ / 128), 256, 0, stream>>>(qkv, xy_bf);  // y_bf reuses x_bf
  gemm_bf16<<<dim3(D_EMB / 128, T_SEQ / 128), blk, 0, stream>>>(
      xy_bf, w_bf, out, T_SEQ, D_EMB, D_EMB);
}
